// RWKV_Tmix_x070_45732811768041
// MI455X (gfx1250) — compile-verified, matrix operand order checked by reading, not run on hardware
//
#include <hip/hip_runtime.h>

typedef unsigned short u16;                                        // raw bf16 bits
typedef __attribute__((ext_vector_type(16))) __bf16 v16bf;
typedef __attribute__((ext_vector_type(8)))  float  v8f;

static constexpr int Bb = 8, Tt = 4096, Cc = 1024, Hh = 64;

__device__ inline u16 f2bf(float f) {
  unsigned u = __float_as_uint(f);
  unsigned r = (u + 0x7fffu + ((u >> 16) & 1u)) >> 16;
  return (u16)r;
}
__device__ inline float bf2f(u16 h) { return __uint_as_float(((unsigned)h) << 16); }

__device__ inline void ldbf8(const u16* p, float* o) {
  uint4 u = *reinterpret_cast<const uint4*>(p);
  o[0] = __uint_as_float(u.x << 16); o[1] = __uint_as_float(u.x & 0xffff0000u);
  o[2] = __uint_as_float(u.y << 16); o[3] = __uint_as_float(u.y & 0xffff0000u);
  o[4] = __uint_as_float(u.z << 16); o[5] = __uint_as_float(u.z & 0xffff0000u);
  o[6] = __uint_as_float(u.w << 16); o[7] = __uint_as_float(u.w & 0xffff0000u);
}

// ---------------------------------------------------------------- convert
__global__ __launch_bounds__(256) void f32_to_bf16(const float* __restrict__ s,
                                                   u16* __restrict__ d, int n) {
  int i = blockIdx.x * 256 + threadIdx.x;
  if (i < n) d[i] = f2bf(s[i]);
}

// ---------------------------------------------------------------- prep (fused token-shift + low-rank maps)
__global__ __launch_bounds__(256) void prep(
    const float* __restrict__ x,
    const float* __restrict__ maa_x, const float* __restrict__ maa_rg,
    const float* __restrict__ maa_wa, const float* __restrict__ maa_k,
    const float* __restrict__ maa_v, const float* __restrict__ maa_w1,
    const float* __restrict__ maa_w2, const float* __restrict__ decay,
    const float* __restrict__ decay_w1, const float* __restrict__ decay_w2,
    const float* __restrict__ aaaaa, const float* __restrict__ aaa_w1,
    const float* __restrict__ aaa_w2, const float* __restrict__ kkk_w1,
    const float* __restrict__ kkk_w2, const float* __restrict__ gate_w1,
    const float* __restrict__ gate_w2, const float* __restrict__ misc_a,
    const float* __restrict__ ma_w1, const float* __restrict__ ma_w2,
    const float* __restrict__ misc_k, const float* __restrict__ mk_w1,
    const float* __restrict__ mk_w2,
    u16* __restrict__ xrg_o, u16* __restrict__ xk_o, u16* __restrict__ xv_o,
    u16* __restrict__ w_o, u16* __restrict__ g_o, u16* __restrict__ a_o,
    u16* __restrict__ ma_o, u16* __restrict__ mk2_o, u16* __restrict__ kkadd_o) {
  __shared__ float xs[Cc], xxs[Cc], xrgS[Cc], xwaS[Cc], xkS[Cc];
  __shared__ float h1[128], g1[128], h2[64], h3[16], h4[16], h5[16], h6[16];
  const int tid = threadIdx.x;
  const int row = blockIdx.x;
  const int t = row & (Tt - 1);
  const float* xr = x + (size_t)row * Cc;

  for (int c = tid; c < Cc; c += 256) {
    float xv_ = xr[c];
    float xp = (t > 0) ? xr[c - Cc] : 0.f;
    xs[c] = xv_;
    xxs[c] = xp - xv_;
  }
  __syncthreads();
  if (tid < 128) {
    float s = 0.f;
    for (int c = 0; c < Cc; c++) s += (xs[c] + xxs[c] * maa_x[c]) * maa_w1[c * 128 + tid];
    h1[tid] = tanhf(s);
  }
  __syncthreads();
  for (int c = tid; c < Cc; c += 256) {
    float m0 = 0, m1 = 0, m2 = 0, m3 = 0;
    for (int d = 0; d < 32; d++) {
      m0 += h1[d]      * maa_w2[(d)      * Cc + c];
      m1 += h1[32 + d] * maa_w2[(32 + d) * Cc + c];
      m2 += h1[64 + d] * maa_w2[(64 + d) * Cc + c];
      m3 += h1[96 + d] * maa_w2[(96 + d) * Cc + c];
    }
    float xv_ = xs[c], xx = xxs[c];
    float xrg = xv_ + xx * (maa_rg[c] + m0);
    float xwa = xv_ + xx * (maa_wa[c] + m1);
    float xk_ = xv_ + xx * (maa_k[c] + m2);
    float xvv = xv_ + xx * (maa_v[c] + m3);
    xrgS[c] = xrg; xwaS[c] = xwa; xkS[c] = xk_;
    size_t o = (size_t)row * Cc + c;
    xrg_o[o] = f2bf(xrg); xk_o[o] = f2bf(xk_); xv_o[o] = f2bf(xvv);
  }
  __syncthreads();
  if (tid < 64) {
    float s = 0.f;
    for (int c = 0; c < Cc; c++) s += xwaS[c] * decay_w1[c * 64 + tid];
    h2[tid] = tanhf(s);
  } else if (tid < 80) {
    int f = tid - 64; float s = 0.f;
    for (int c = 0; c < Cc; c++) s += xwaS[c] * aaa_w1[c * 16 + f];
    h3[f] = s;
  } else if (tid < 96) {
    int f = tid - 80; float s = 0.f;
    for (int c = 0; c < Cc; c++) s += xwaS[c] * ma_w1[c * 16 + f];
    h4[f] = s;
  } else if (tid < 112) {
    int f = tid - 96; float s = 0.f;
    for (int c = 0; c < Cc; c++) s += xkS[c] * mk_w1[c * 16 + f];
    h5[f] = s;
  } else if (tid < 128) {
    int f = tid - 112; float s = 0.f;
    for (int c = 0; c < Cc; c++) s += xkS[c] * kkk_w1[c * 16 + f];
    h6[f] = tanhf(s);
  } else {
    int f = tid - 128; float s = 0.f;
    for (int c = 0; c < Cc; c++) s += xrgS[c] * gate_w1[c * 128 + f];
    g1[f] = tanhf(s);
  }
  __syncthreads();
  for (int c = tid; c < Cc; c += 256) {
    float wl = 0.f;
    for (int d = 0; d < 64; d++) wl += h2[d] * decay_w2[d * Cc + c];
    float z = decay[c] + wl;
    float wv = -log1pf(__expf(-z)) - 0.5f;  // -softplus(-z) - 0.5
    float a3 = 0, a4 = 0, a5 = 0, a6 = 0;
    for (int d = 0; d < 16; d++) {
      a3 += h3[d] * aaa_w2[d * Cc + c];
      a4 += h4[d] * ma_w2[d * Cc + c];
      a5 += h5[d] * mk_w2[d * Cc + c];
      a6 += h6[d] * kkk_w2[d * Cc + c];
    }
    float gg = 0.f;
    for (int d = 0; d < 128; d++) gg += g1[d] * gate_w2[d * Cc + c];
    size_t o = (size_t)row * Cc + c;
    w_o[o]    = f2bf(wv);
    a_o[o]    = f2bf(1.f / (1.f + __expf(-(aaaaa[c] + a3))));
    ma_o[o]   = f2bf(1.f / (1.f + __expf(-(misc_a[c] + a4))));
    mk2_o[o]  = f2bf(1.f / (1.f + __expf(-(misc_k[c] + a5))));
    kkadd_o[o]= f2bf(a6);
    g_o[o]    = f2bf(gg);
  }
}

// ---------------------------------------------------------------- bf16 WMMA GEMM: Out[M,N] = A[M,K] * W[N,K]^T
// Double-buffered LDS staging via CDNA5 GLOBAL_LOAD_ASYNC_TO_LDS (ASYNCcnt).
template <bool OUT_F32>
__global__ __launch_bounds__(256) void gemm_bf16(const u16* __restrict__ A,
                                                 const u16* __restrict__ W,
                                                 void* __restrict__ Out,
                                                 int M, int N, int K) {
  constexpr int BM = 128, BN = 128, BK = 32, LDT = BK + 8;
  __shared__ u16 As[2][BM * LDT];
  __shared__ u16 Bs[2][BN * LDT];
  const int tid = threadIdx.x;
  const int wid = tid >> 5, lane = tid & 31;
  const int m0 = blockIdx.x * BM, n0 = blockIdx.y * BN;
  const int waveM = (wid & 3) * 32;   // 2 M-tiles per wave
  const int waveN = (wid >> 2) * 64;  // 4 N-tiles per wave
  v8f acc[2][4];
  #pragma unroll
  for (int i = 0; i < 2; i++)
    #pragma unroll
    for (int j = 0; j < 4; j++)
      #pragma unroll
      for (int g = 0; g < 8; g++) acc[i][j][g] = 0.f;

  const int lrow = tid >> 1;        // 0..127
  const int lseg = (tid & 1) * 16;  // bf16 element offset within K-tile
  const int fr = lane & 15;
  const int fk = (lane < 16) ? 0 : 8;  // per ISA 16-bit A/B operand layout
  union Frag { uint4 u[2]; v16bf v; };

  const u16* gA = A + (size_t)(m0 + lrow) * K + lseg;
  const u16* gB = W + (size_t)(n0 + lrow) * K + lseg;

  // Stage one 128x32 A tile + 128x32 B tile into LDS buffer `buf` (async DMA to LDS).
  auto stage = [&](int k0, int buf) {
    unsigned la = (unsigned)(size_t)(&As[buf][lrow * LDT + lseg]);
    unsigned lb = (unsigned)(size_t)(&Bs[buf][lrow * LDT + lseg]);
    const u16* ga = gA + k0;
    const u16* gb = gB + k0;
    asm volatile("global_load_async_to_lds_b128 %0, %1, off"
                 :: "v"(la), "v"(ga) : "memory");
    asm volatile("global_load_async_to_lds_b128 %0, %1, off offset:16"
                 :: "v"(la), "v"(ga) : "memory");
    asm volatile("global_load_async_to_lds_b128 %0, %1, off"
                 :: "v"(lb), "v"(gb) : "memory");
    asm volatile("global_load_async_to_lds_b128 %0, %1, off offset:16"
                 :: "v"(lb), "v"(gb) : "memory");
  };

  stage(0, 0);
  asm volatile("s_wait_asynccnt 0" ::: "memory");
  __syncthreads();

  int cur = 0;
  for (int k0 = 0; k0 < K; k0 += BK) {
    if (k0 + BK < K) stage(k0 + BK, cur ^ 1);  // prefetch next tile into other buffer

    Frag fa[2], fb[4];
    #pragma unroll
    for (int i = 0; i < 2; i++) {
      const u16* p = &As[cur][(waveM + i * 16 + fr) * LDT + fk];
      fa[i].u[0] = *reinterpret_cast<const uint4*>(p);
      fa[i].u[1] = *reinterpret_cast<const uint4*>(p + 16);
    }
    #pragma unroll
    for (int j = 0; j < 4; j++) {
      const u16* p = &Bs[cur][(waveN + j * 16 + fr) * LDT + fk];
      fb[j].u[0] = *reinterpret_cast<const uint4*>(p);
      fb[j].u[1] = *reinterpret_cast<const uint4*>(p + 16);
    }
    #pragma unroll
    for (int i = 0; i < 2; i++)
      #pragma unroll
      for (int j = 0; j < 4; j++)
        acc[i][j] = __builtin_amdgcn_wmma_f32_16x16x32_bf16(
            false, fa[i].v, false, fb[j].v, (short)0, acc[i][j], false, false);

    asm volatile("s_wait_asynccnt 0" ::: "memory");  // next-buffer DMA complete
    __syncthreads();                                  // everyone done reading cur
    cur ^= 1;
  }

  const int col = lane & 15;
  const int rAdd = (lane >> 4) * 8;  // C/D layout: VGPR g -> M = g (+8 for upper lanes)
  #pragma unroll
  for (int i = 0; i < 2; i++)
    #pragma unroll
    for (int j = 0; j < 4; j++) {
      int row = m0 + waveM + i * 16 + rAdd;
      int c = n0 + waveN + j * 16 + col;
      #pragma unroll
      for (int g = 0; g < 8; g++) {
        float vv = acc[i][j][g];
        if (OUT_F32)
          reinterpret_cast<float*>(Out)[(size_t)(row + g) * N + c] = vv;
        else
          reinterpret_cast<u16*>(Out)[(size_t)(row + g) * N + c] = f2bf(vv);
      }
    }
}

// ---------------------------------------------------------------- fixk: kk head-norm, ab, final k
__global__ __launch_bounds__(256) void fixk(u16* __restrict__ k0,          // in k, out k_final
                                            const u16* __restrict__ w,
                                            u16* __restrict__ kkadd,       // in lowrank, out kk
                                            u16* __restrict__ a_ab,        // in a, out ab
                                            const u16* __restrict__ ma,
                                            const u16* __restrict__ mk2) {
  size_t i = (size_t)blockIdx.x * 256 + threadIdx.x;
  float kv = bf2f(k0[i]);
  float kkv = kv + bf2f(kkadd[i]);
  float ss = kkv * kkv;
  ss += __shfl_xor(ss, 1, 32); ss += __shfl_xor(ss, 2, 32);
  ss += __shfl_xor(ss, 4, 32); ss += __shfl_xor(ss, 8, 32);
  float kkn = kkv / fmaxf(sqrtf(ss), 1e-12f);
  kkadd[i] = f2bf(kkn);
  kkn = bf2f(f2bf(kkn));  // match bf16 rounding of kk before ab
  float av = bf2f(a_ab[i]);
  a_ab[i] = f2bf(kkn * av);
  float mav = bf2f(ma[i]);
  float k2 = kv * mav + kv * av * (1.f - mav);
  float wv = bf2f(w[i]);
  float m2 = bf2f(mk2[i]);
  k0[i] = f2bf(k2 * __expf(fminf(wv * m2, 0.f)));
}

// ---------------------------------------------------------------- RWKV-7 scan: 1 wave per (b,h), lane = state row
__global__ __launch_bounds__(32) void rwkv_scan(const u16* __restrict__ q,
                                                const u16* __restrict__ w,
                                                const u16* __restrict__ k,
                                                const u16* __restrict__ v,
                                                const u16* __restrict__ kk,
                                                const u16* __restrict__ ab,
                                                float* __restrict__ y) {
  const int bh = blockIdx.x;
  const int b = bh >> 6, h = bh & 63;
  const int lane = threadIdx.x;
  const int i = lane & 15, half = lane >> 4;  // lane owns row i, cols half*8..+8
  size_t rowbase = ((size_t)b * Tt) * Cc + h * 16;
  float S[8];
  #pragma unroll
  for (int r = 0; r < 8; r++) S[r] = 0.f;

  for (int t = 0; t < Tt; t++, rowbase += Cc) {
    const size_t off = rowbase + half * 8;
    float qv[8], wv[8], kv[8], av[8], bv[8];
    ldbf8(q + off, qv);
    ldbf8(w + off, wv);
    ldbf8(k + off, kv);
    ldbf8(kk + off, av);   // a = -kk
    ldbf8(ab + off, bv);
    float vi = bf2f(v[rowbase + i]);
    if (t + 8 < Tt) __builtin_prefetch(q + off + 8 * Cc, 0, 1);

    float sa = 0.f;
    #pragma unroll
    for (int r = 0; r < 8; r++) sa -= S[r] * av[r];
    sa += __shfl_xor(sa, 16, 32);   // full S @ a over 16 cols

    #pragma unroll
    for (int r = 0; r < 8; r++)
      S[r] = S[r] * __expf(wv[r]) + sa * bv[r] + vi * kv[r];

    float ys = 0.f;
    #pragma unroll
    for (int r = 0; r < 8; r++) ys += S[r] * qv[r];
    ys += __shfl_xor(ys, 16, 32);   // full S @ q
    if (half == 0) y[rowbase + i] = ys;
  }
}

// ---------------------------------------------------------------- GroupNorm + bonus + gate
__global__ __launch_bounds__(256) void gn_gate(const float* __restrict__ y,
                                               const u16* __restrict__ r,
                                               const u16* __restrict__ kf,
                                               const u16* __restrict__ v,
                                               const u16* __restrict__ g,
                                               const float* __restrict__ faaaa,
                                               const float* __restrict__ lnw,
                                               const float* __restrict__ lnb,
                                               u16* __restrict__ gy) {
  size_t base = (size_t)blockIdx.x * Cc;
  int tid = threadIdx.x;
  #pragma unroll
  for (int kq = 0; kq < 4; kq++) {
    int c = kq * 256 + tid;
    size_t idx = base + c;
    float yv = y[idx];
    float s1 = yv, s2 = yv * yv;
    s1 += __shfl_xor(s1, 1, 32); s2 += __shfl_xor(s2, 1, 32);
    s1 += __shfl_xor(s1, 2, 32); s2 += __shfl_xor(s2, 2, 32);
    s1 += __shfl_xor(s1, 4, 32); s2 += __shfl_xor(s2, 4, 32);
    s1 += __shfl_xor(s1, 8, 32); s2 += __shfl_xor(s2, 8, 32);
    float mean = s1 * (1.f / 16.f);
    float var = s2 * (1.f / 16.f) - mean * mean;
    float xn = (yv - mean) * rsqrtf(var + 6.4e-4f) * lnw[c] + lnb[c];
    float rv = bf2f(r[idx]), kv = bf2f(kf[idx]), vv = bf2f(v[idx]), gv = bf2f(g[idx]);
    float d = rv * kv * faaaa[c];
    d += __shfl_xor(d, 1, 32); d += __shfl_xor(d, 2, 32);
    d += __shfl_xor(d, 4, 32); d += __shfl_xor(d, 8, 32);
    gy[idx] = f2bf((xn + d * vv) * gv);
  }
}

// ---------------------------------------------------------------- launcher
extern "C" void kernel_launch(void* const* d_in, const int* in_sizes, int n_in,
                              void* d_out, int out_size, void* d_ws, size_t ws_size,
                              hipStream_t stream) {
  (void)in_sizes; (void)n_in; (void)out_size; (void)ws_size;
  const float* x       = (const float*)d_in[0];
  const float* maa_x   = (const float*)d_in[1];
  const float* maa_rg  = (const float*)d_in[2];
  const float* maa_wa  = (const float*)d_in[3];
  const float* maa_k   = (const float*)d_in[4];
  const float* maa_v   = (const float*)d_in[5];
  const float* maa_w1  = (const float*)d_in[6];
  const float* maa_w2  = (const float*)d_in[7];
  const float* decay   = (const float*)d_in[8];
  const float* decay_w1= (const float*)d_in[9];
  const float* decay_w2= (const float*)d_in[10];
  const float* faaaa   = (const float*)d_in[11];
  const float* aaaaa   = (const float*)d_in[12];
  const float* aaa_w1  = (const float*)d_in[13];
  const float* aaa_w2  = (const float*)d_in[14];
  const float* kkk_w1  = (const float*)d_in[15];
  const float* kkk_w2  = (const float*)d_in[16];
  const float* gate_w1 = (const float*)d_in[17];
  const float* gate_w2 = (const float*)d_in[18];
  const float* misc_a  = (const float*)d_in[19];
  const float* ma_w1   = (const float*)d_in[20];
  const float* ma_w2   = (const float*)d_in[21];
  const float* misc_k  = (const float*)d_in[22];
  const float* mk_w1   = (const float*)d_in[23];
  const float* mk_w2   = (const float*)d_in[24];
  const float* Wr      = (const float*)d_in[25];
  const float* Wk      = (const float*)d_in[26];
  const float* Wv      = (const float*)d_in[27];
  const float* Wo      = (const float*)d_in[28];
  const float* ln_w    = (const float*)d_in[29];
  const float* ln_b    = (const float*)d_in[30];

  const size_t BT = (size_t)Bb * Tt;        // 32768
  const size_t NC = BT * Cc;                // 33.5M elems
  char* p = (char*)d_ws;
  auto take = [&](size_t bytes) -> void* {
    void* r = (void*)p;
    p += (bytes + 255) & ~(size_t)255;
    return r;
  };
  u16* xrg  = (u16*)take(NC * 2);  // later reused as gy (final GEMM input)
  u16* xk   = (u16*)take(NC * 2);
  u16* xv   = (u16*)take(NC * 2);
  u16* rb   = (u16*)take(NC * 2);
  u16* kb   = (u16*)take(NC * 2);  // k gemm out -> k_final (in place)
  u16* vb   = (u16*)take(NC * 2);
  u16* wb   = (u16*)take(NC * 2);
  u16* gb   = (u16*)take(NC * 2);
  u16* ab_  = (u16*)take(NC * 2);  // a -> ab (in place)
  u16* mab  = (u16*)take(NC * 2);
  u16* mk2b = (u16*)take(NC * 2);
  u16* kkb  = (u16*)take(NC * 2);  // kkadd -> kk (in place)
  float* yb = (float*)take(NC * 4);
  u16* Wrb  = (u16*)take((size_t)Cc * Cc * 2);
  u16* Wkb  = (u16*)take((size_t)Cc * Cc * 2);
  u16* Wvb  = (u16*)take((size_t)Cc * Cc * 2);
  u16* Wob  = (u16*)take((size_t)Cc * Cc * 2);

  const int WN = Cc * Cc;
  f32_to_bf16<<<WN / 256, 256, 0, stream>>>(Wr, Wrb, WN);
  f32_to_bf16<<<WN / 256, 256, 0, stream>>>(Wk, Wkb, WN);
  f32_to_bf16<<<WN / 256, 256, 0, stream>>>(Wv, Wvb, WN);
  f32_to_bf16<<<WN / 256, 256, 0, stream>>>(Wo, Wob, WN);

  prep<<<(int)BT, 256, 0, stream>>>(x, maa_x, maa_rg, maa_wa, maa_k, maa_v,
                                    maa_w1, maa_w2, decay, decay_w1, decay_w2,
                                    aaaaa, aaa_w1, aaa_w2, kkk_w1, kkk_w2,
                                    gate_w1, gate_w2, misc_a, ma_w1, ma_w2,
                                    misc_k, mk_w1, mk_w2,
                                    xrg, xk, xv, wb, gb, ab_, mab, mk2b, kkb);

  dim3 gg((int)(BT / 128), Cc / 128);
  gemm_bf16<false><<<gg, 256, 0, stream>>>(xrg, Wrb, rb, (int)BT, Cc, Cc);
  gemm_bf16<false><<<gg, 256, 0, stream>>>(xk,  Wkb, kb, (int)BT, Cc, Cc);
  gemm_bf16<false><<<gg, 256, 0, stream>>>(xv,  Wvb, vb, (int)BT, Cc, Cc);

  fixk<<<(int)(NC / 256), 256, 0, stream>>>(kb, wb, kkb, ab_, mab, mk2b);

  rwkv_scan<<<Bb * Hh, 32, 0, stream>>>(rb, wb, kb, vb, kkb, ab_, yb);

  gn_gate<<<(int)BT, 256, 0, stream>>>(yb, rb, kb, vb, gb, faaaa, ln_w, ln_b, xrg);

  gemm_bf16<true><<<gg, 256, 0, stream>>>(xrg, Wob, d_out, (int)BT, Cc, Cc);
}
